// MyModel_61933428412569
// MI455X (gfx1250) — matrix operand order repaired and checked by reading, not compiled, not run on hardware
//
#include <hip/hip_runtime.h>

typedef _Float16 v8h  __attribute__((ext_vector_type(8)));
typedef _Float16 v16h __attribute__((ext_vector_type(16)));
typedef float    v8f  __attribute__((ext_vector_type(8)));

#define THREADS 256
#define ROWS_PER_BLOCK 256   // 8 waves x 32 rows (2 M-tiles per wave)

// ---- LDS byte offsets (16B aligned). Phase-based aliasing noted. ----
#define OFF_XH    0u        // 256 x 200 f16 = 102400  (x, f16, stride 200)
#define OFF_WA    102400u   // 192 x 200 f16 = 76800   (Wih_a rows {0..63,128..255})
#define OFF_WC    179200u   // 192 x 200 f16 = 76800   (Wih_c same)
#define OFF_C1H   256000u   // 64 x 104 f16  = 13312   (W_c1^T padded K=96)
#define OFF_BIASA 269312u   // 256 f32
#define OFF_BIASC 270336u   // 256 f32
#define OFF_WACT  271360u   // 128 f32
#define OFF_BACT  271872u   // 2 f32 (pad 64)
#define OFF_WC2   271936u   // 64 f32
#define OFF_BC2   272192u   // 1 f32 (pad 64)
#define OFF_BC1   272256u   // 64 f32
#define LDS_BYTES 272512u   // ~266 KB <= 320 KB / WGP
// aliases (different phases, separated by barriers):
#define OFF_HA    OFF_WA    // 256 x 65 f32 = 66560 <= 76800 (after gemm_a reads done)
#define OFF_COMB  OFF_XH    // 256 x 104 f16 = 53248 <= 102400 (after gemm_c reads done)
#define OFF_ZB    OFF_WC    // 256 x 65 f32 = 66560 <= 76800 (after gemm_c reads done)

__device__ __forceinline__ float sigmoid_f(float x) {
    float e = __builtin_amdgcn_exp2f(x * -1.44269504f);   // v_exp_f32
    return __builtin_amdgcn_rcpf(1.0f + e);               // v_rcp_f32
}
__device__ __forceinline__ float tanh_f(float x) {
    float xc = fminf(fmaxf(x, -16.0f), 16.0f);
    float e = __builtin_amdgcn_exp2f(xc * 2.88539008f);   // e^(2x)
    return (e - 1.0f) * __builtin_amdgcn_rcpf(e + 1.0f);
}

__global__ __launch_bounds__(THREADS, 1)
void fused_actor_critic(const float* __restrict__ state,
                        const float* __restrict__ Wih_a,
                        const float* __restrict__ bih_a, const float* __restrict__ bhh_a,
                        const float* __restrict__ W_act, const float* __restrict__ b_act,
                        const float* __restrict__ Wih_c,
                        const float* __restrict__ bih_c, const float* __restrict__ bhh_c,
                        const float* __restrict__ W_c1,  const float* __restrict__ b_c1,
                        const float* __restrict__ W_c2,  const float* __restrict__ b_c2,
                        float* __restrict__ out)
{
    extern __shared__ char smem[];
    _Float16* xh    = (_Float16*)(smem + OFF_XH);
    _Float16* wa    = (_Float16*)(smem + OFF_WA);
    _Float16* wc    = (_Float16*)(smem + OFF_WC);
    _Float16* c1h   = (_Float16*)(smem + OFF_C1H);
    float*    biasa = (float*)(smem + OFF_BIASA);
    float*    biasc = (float*)(smem + OFF_BIASC);
    float*    wact  = (float*)(smem + OFF_WACT);
    float*    bact  = (float*)(smem + OFF_BACT);
    float*    wc2   = (float*)(smem + OFF_WC2);
    float*    bc2   = (float*)(smem + OFF_BC2);
    float*    bc1   = (float*)(smem + OFF_BC1);
    float*    ha    = (float*)(smem + OFF_HA);
    _Float16* comb  = (_Float16*)(smem + OFF_COMB);
    float*    zb    = (float*)(smem + OFF_ZB);

    const int tid = threadIdx.x;
    const long long rowblk = (long long)blockIdx.x * ROWS_PER_BLOCK;

    // ---------------- stage to LDS (f32 -> f16 WMMA operands) ----------------
    for (int i = tid; i < ROWS_PER_BLOCK * 192; i += THREADS) {
        int r = i / 192, c = i % 192;
        xh[r * 200 + c] = (_Float16)state[(rowblk + r) * 192 + c];
    }
    // only gate rows i(0..63), g(128..191), o(192..255): 192 staged rows
    for (int i = tid; i < 192 * 192; i += THREADS) {
        int sr = i / 192, c = i % 192;
        int gr = (sr < 64) ? sr : sr + 64;
        wa[sr * 200 + c] = (_Float16)Wih_a[gr * 192 + c];
        wc[sr * 200 + c] = (_Float16)Wih_c[gr * 192 + c];
    }
    // critic B matrix: c1h[n][k] = W_c1[n][k] for k<66, else 0 (K padded to 96)
    for (int i = tid; i < 64 * 96; i += THREADS) {
        int n = i / 96, k = i % 96;
        c1h[n * 104 + k] = (_Float16)((k < 66) ? W_c1[n * 66 + k] : 0.0f);
    }
    biasa[tid] = bih_a[tid] + bhh_a[tid];
    biasc[tid] = bih_c[tid] + bhh_c[tid];
    if (tid < 128) wact[tid] = W_act[tid];
    if (tid < 2)   bact[tid] = b_act[tid];
    if (tid < 64)  { wc2[tid] = W_c2[tid]; bc1[tid] = b_c1[tid]; }
    if (tid == 0)  bc2[0] = b_c2[0];
    __syncthreads();                                   // [B1]

    const int lane  = tid & 31;
    const int wave  = tid >> 5;
    const int rb    = wave * 32;            // this wave's row base (2 M-tiles)
    const int lsel  = lane & 15;
    const int khalf = (lane >> 4) * 8;      // A-frag: K 0..7 / 8..15 (+16 for 2nd half)
    const int kbsel = (lane >> 4) * 16;     // unused; B-frag reads K via khalf, same order as A
    const int rofs  = (lane >> 4) << 3;     // C layout: lanes 16-31 hold rows M+8

    v8f acc0[12], acc1[12];                 // M-tiles 0/1 x N-tiles {i0..3,g0..3,o0..3}

    // gates = x @ Wih^T (i,g,o tiles only), K=192 in 6 steps; B double-buffered,
    // each B fragment reused by two WMMAs (M-blocking) to halve DS traffic.
    auto run_gemm = [&](const _Float16* W) {
        #pragma unroll
        for (int t = 0; t < 12; ++t)
            #pragma unroll
            for (int j = 0; j < 8; ++j) { acc0[t][j] = 0.0f; acc1[t][j] = 0.0f; }
        #pragma unroll
        for (int kt = 0; kt < 6; ++kt) {
            v16h af0, af1;
            const _Float16* ap = xh + (rb + lsel) * 200 + kt * 32 + khalf;
            ((v8h*)&af0)[0] = *(const v8h*)(ap);
            ((v8h*)&af0)[1] = *(const v8h*)(ap + 16);
            ((v8h*)&af1)[0] = *(const v8h*)(ap + 16 * 200);
            ((v8h*)&af1)[1] = *(const v8h*)(ap + 16 * 200 + 16);
            v16h bf[2];
            {
                const _Float16* bp = W + lsel * 200 + kt * 32 + khalf;
                ((v8h*)&bf[0])[0] = *(const v8h*)(bp);
                ((v8h*)&bf[0])[1] = *(const v8h*)(bp + 16);
            }
            #pragma unroll
            for (int t = 0; t < 12; ++t) {
                if (t < 11) {   // prefetch next B fragment into the other buffer
                    const _Float16* bp = W + ((t + 1) * 16 + lsel) * 200 + kt * 32 + khalf;
                    ((v8h*)&bf[(t + 1) & 1])[0] = *(const v8h*)(bp);
                    ((v8h*)&bf[(t + 1) & 1])[1] = *(const v8h*)(bp + 16);
                }
                acc0[t] = __builtin_amdgcn_wmma_f32_16x16x32_f16(
                    false, af0, false, bf[t & 1], (short)0, acc0[t], false, false);
                acc1[t] = __builtin_amdgcn_wmma_f32_16x16x32_f16(
                    false, af1, false, bf[t & 1], (short)0, acc1[t], false, false);
            }
        }
    };

    // ---------------- actor LSTM ----------------
    run_gemm(wa);
    __syncthreads();                                   // [B2] wa reads done -> ha may alias
    #pragma unroll
    for (int n = 0; n < 4; ++n) {
        float bi = biasa[      n * 16 + lsel];
        float bg = biasa[128 + n * 16 + lsel];
        float bo = biasa[192 + n * 16 + lsel];
        #pragma unroll
        for (int m = 0; m < 2; ++m) {
            #pragma unroll
            for (int j = 0; j < 8; ++j) {
                float gi = (m ? acc1[n][j]     : acc0[n][j])     + bi;
                float gg = (m ? acc1[4 + n][j] : acc0[4 + n][j]) + bg;
                float go = (m ? acc1[8 + n][j] : acc0[8 + n][j]) + bo;
                float h  = sigmoid_f(go) * tanh_f(sigmoid_f(gi) * tanh_f(gg)); // c0=0
                ha[(rb + m * 16 + j + rofs) * 65 + n * 16 + lsel] = h;
            }
        }
    }

    // ---------------- critic LSTM ----------------
    run_gemm(wc);
    __syncthreads();                                   // [B3] xh reads done -> comb may alias
    #pragma unroll
    for (int n = 0; n < 4; ++n) {
        float bi = biasc[      n * 16 + lsel];
        float bg = biasc[128 + n * 16 + lsel];
        float bo = biasc[192 + n * 16 + lsel];
        #pragma unroll
        for (int m = 0; m < 2; ++m) {
            #pragma unroll
            for (int j = 0; j < 8; ++j) {
                float gi = (m ? acc1[n][j]     : acc0[n][j])     + bi;
                float gg = (m ? acc1[4 + n][j] : acc0[4 + n][j]) + bg;
                float go = (m ? acc1[8 + n][j] : acc0[8 + n][j]) + bo;
                float h  = sigmoid_f(go) * tanh_f(sigmoid_f(gi) * tanh_f(gg));
                comb[(rb + m * 16 + j + rofs) * 104 + n * 16 + lsel] = (_Float16)h;
            }
        }
    }
    __syncthreads();                                   // [B4] ha + comb[h_c] complete

    // ---------------- actor head (1 thread per row, both outputs) ----------------
    {
        int row = tid;
        float a0 = bact[0], a1 = bact[1];
        #pragma unroll 8
        for (int h = 0; h < 64; ++h) {
            float hv = ha[row * 65 + h];
            a0 += hv * wact[h];
            a1 += hv * wact[64 + h];
        }
        comb[row * 104 + 64] = (_Float16)a0;
        comb[row * 104 + 65] = (_Float16)a1;
        for (int k = 66; k < 96; ++k)               // zero K-padding
            comb[row * 104 + k] = (_Float16)0.0f;
    }
    __syncthreads();                                   // [B5] combined ready

    // ---------------- critic layer1: z = relu(combined @ W_c1^T + b_c1) ----------------
    v8f zacc0[4], zacc1[4];
    #pragma unroll
    for (int t = 0; t < 4; ++t)
        #pragma unroll
        for (int j = 0; j < 8; ++j) { zacc0[t][j] = 0.0f; zacc1[t][j] = 0.0f; }
    #pragma unroll
    for (int kt = 0; kt < 3; ++kt) {
        v16h af0, af1;
        const _Float16* ap = comb + (rb + lsel) * 104 + kt * 32 + khalf;
        ((v8h*)&af0)[0] = *(const v8h*)(ap);
        ((v8h*)&af0)[1] = *(const v8h*)(ap + 16);
        ((v8h*)&af1)[0] = *(const v8h*)(ap + 16 * 104);
        ((v8h*)&af1)[1] = *(const v8h*)(ap + 16 * 104 + 16);
        v16h bf[2];
        {
            const _Float16* bp = c1h + lsel * 104 + kt * 32 + khalf;
            ((v8h*)&bf[0])[0] = *(const v8h*)(bp);
            ((v8h*)&bf[0])[1] = *(const v8h*)(bp + 16);
        }
        #pragma unroll
        for (int t = 0; t < 4; ++t) {
            if (t < 3) {
                const _Float16* bp = c1h + ((t + 1) * 16 + lsel) * 104 + kt * 32 + khalf;
                ((v8h*)&bf[(t + 1) & 1])[0] = *(const v8h*)(bp);
                ((v8h*)&bf[(t + 1) & 1])[1] = *(const v8h*)(bp + 16);
            }
            zacc0[t] = __builtin_amdgcn_wmma_f32_16x16x32_f16(
                false, af0, false, bf[t & 1], (short)0, zacc0[t], false, false);
            zacc1[t] = __builtin_amdgcn_wmma_f32_16x16x32_f16(
                false, af1, false, bf[t & 1], (short)0, zacc1[t], false, false);
        }
    }
    #pragma unroll
    for (int n = 0; n < 4; ++n) {
        float bz = bc1[n * 16 + lsel];
        #pragma unroll
        for (int m = 0; m < 2; ++m)
            #pragma unroll
            for (int j = 0; j < 8; ++j) {
                float z = (m ? zacc1[n][j] : zacc0[n][j]) + bz;
                zb[(rb + m * 16 + j + rofs) * 65 + n * 16 + lsel] = fmaxf(z, 0.0f);
            }
    }
    __syncthreads();                                   // [B6]

    // ---------------- critic layer2: out = z @ W_c2^T + b_c2 (1 thread/row) ----------------
    {
        float v = bc2[0];
        #pragma unroll 8
        for (int k = 0; k < 64; ++k)
            v += zb[tid * 65 + k] * wc2[k];
        out[rowblk + tid] = v;
    }
}

extern "C" void kernel_launch(void* const* d_in, const int* in_sizes, int n_in,
                              void* d_out, int out_size, void* d_ws, size_t ws_size,
                              hipStream_t stream) {
    (void)n_in; (void)d_ws; (void)ws_size; (void)out_size;
    const float* state = (const float*)d_in[0];
    const float* Wih_a = (const float*)d_in[1];
    // d_in[2] = Whh_a: unused (h0 == 0)
    const float* bih_a = (const float*)d_in[3];
    const float* bhh_a = (const float*)d_in[4];
    const float* W_act = (const float*)d_in[5];
    const float* b_act = (const float*)d_in[6];
    const float* Wih_c = (const float*)d_in[7];
    // d_in[8] = Whh_c: unused
    const float* bih_c = (const float*)d_in[9];
    const float* bhh_c = (const float*)d_in[10];
    const float* W_c1  = (const float*)d_in[11];
    const float* b_c1  = (const float*)d_in[12];
    const float* W_c2  = (const float*)d_in[13];
    const float* b_c2  = (const float*)d_in[14];
    float* out = (float*)d_out;

    int B = in_sizes[0] / 192;
    int blocks = B / ROWS_PER_BLOCK;

    hipFuncSetAttribute((const void*)fused_actor_critic,
                        hipFuncAttributeMaxDynamicSharedMemorySize, (int)LDS_BYTES);

    fused_actor_critic<<<blocks, THREADS, LDS_BYTES, stream>>>(
        state, Wih_a, bih_a, bhh_a, W_act, b_act,
        Wih_c, bih_c, bhh_c, W_c1, b_c1, W_c2, b_c2, out);
}
